// UnifiedModel_35588099014951
// MI455X (gfx1250) — matrix operand order repaired and checked by reading, not compiled, not run on hardware
//
#include <hip/hip_runtime.h>

typedef __attribute__((ext_vector_type(16))) __bf16 v16bf;
typedef __attribute__((ext_vector_type(8)))  __bf16 v8bf;
typedef __attribute__((ext_vector_type(8)))  float  v8f;

#define HD   256
#define FFD  512
#define NB   16
#define SL   2048
#define VOC  32000
#define CCH  32    // WY chunk length for the delta-rule scan

// ---------------------------------------------------------------------------
// WMMA helpers (bf16 in, f32 accumulate), CDNA5 16x16x32 layout
// ---------------------------------------------------------------------------
__device__ __forceinline__ v16bf wmma_load_a(const __bf16* base, int stride, int k0, int lane) {
  // A 16x32 (MxK) bf16: lane row = lane&15; lanes<16 -> K chunks {0..7},{16..23};
  // lanes>=16 -> {8..15},{24..31}. Two contiguous 16B loads.
  const int row = lane & 15;
  const int c0  = (lane < 16) ? 0 : 8;
  const __bf16* p = base + row * stride + k0 + c0;
  v8bf lo = *(const v8bf*)(p);
  v8bf hi = *(const v8bf*)(p + 16);
  v16bf a;
#pragma unroll
  for (int i = 0; i < 8; ++i) { a[i] = lo[i]; a[i + 8] = hi[i]; }
  return a;
}

// Same A layout but sourced from an fp32 matrix (convert on the fly).
__device__ __forceinline__ v16bf wmma_load_a_f32(const float* base, int stride, int k0, int lane) {
  const int row = lane & 15;
  const int c0  = (lane < 16) ? 0 : 8;
  const float* p = base + row * stride + k0 + c0;
  v16bf a;
#pragma unroll
  for (int i = 0; i < 8; ++i) { a[i] = (__bf16)p[i]; a[i + 8] = (__bf16)p[16 + i]; }
  return a;
}

__device__ __forceinline__ v16bf wmma_load_b(const __bf16* wt_nk, int K, int n0, int k0, int lane) {
  // B 32x16 (KxN) bf16 from (N,K)-major storage: lane col = lane&15;
  // lanes<16 -> K chunks {0..7},{16..23}; lanes>=16 -> {8..15},{24..31}.
  const int col = lane & 15;
  const int kb  = k0 + ((lane < 16) ? 0 : 8);
  const __bf16* p = wt_nk + (size_t)(n0 + col) * K + kb;
  v8bf lo = *(const v8bf*)(p);
  v8bf hi = *(const v8bf*)(p + 16);
  v16bf b;
#pragma unroll
  for (int i = 0; i < 8; ++i) { b[i] = lo[i]; b[i + 8] = hi[i]; }
  return b;
}

__device__ __forceinline__ v8f wmma_bf16(v16bf a, v16bf b, v8f c) {
  return __builtin_amdgcn_wmma_f32_16x16x32_bf16(false, a, false, b, (short)0, c, false, false);
}

// ---------------------------------------------------------------------------
// Weight convert + transpose: out[n*K + k] = bf16(in[k*N + n])
// ---------------------------------------------------------------------------
__global__ void conv_transpose_bf16(const float* __restrict__ in, __bf16* __restrict__ out,
                                    int K, int N) {
  size_t total = (size_t)K * N;
  for (size_t i = (size_t)blockIdx.x * blockDim.x + threadIdx.x; i < total;
       i += (size_t)gridDim.x * blockDim.x) {
    size_t n = i / (size_t)K;
    size_t k = i - n * (size_t)K;
    out[i] = (__bf16)in[k * (size_t)N + n];
  }
}

// ---------------------------------------------------------------------------
// Fused: embed gather -> FFN (relu(e@w1+b1)@w2+b2) -> residual -> LayerNorm
//        -> k = h @ kp_w.   One block = 16 rows, 256 threads (8 waves).
// ---------------------------------------------------------------------------
__global__ void __launch_bounds__(256)
ffn_ln_kproj(const int* __restrict__ seq, const float* __restrict__ embed,
             const __bf16* __restrict__ w1t, const float* __restrict__ b1,
             const __bf16* __restrict__ w2t, const float* __restrict__ b2,
             const float* __restrict__ ln_g, const float* __restrict__ ln_b,
             const __bf16* __restrict__ kpt, float* __restrict__ k_all) {
  extern __shared__ char smem[];
  float*  e_f32 = (float*)smem;                    // 16*256 f32 (residual)
  float*  x_f32 = e_f32 + 16 * HD;                 // 16*256 f32 (pre-LN)
  __bf16* e_bf  = (__bf16*)(x_f32 + 16 * HD);      // 16*256 bf16 (A for GEMM1)
  __bf16* h1_bf = e_bf + 16 * HD;                  // 16*512 bf16 (A for GEMM2)
  __bf16* h_bf  = h1_bf + 16 * FFD;                // 16*256 bf16 (A for GEMM3)

  const int tid  = threadIdx.x;
  const int lane = tid & 31;
  const int wv   = tid >> 5;
  const int rowbase = blockIdx.x * 16;

  // ---- gather embedding rows ----
  for (int i = tid; i < 16 * HD; i += 256) {
    int r = i >> 8, c = i & 255;
    int tok = seq[rowbase + r];
    float v = embed[(size_t)tok * HD + c];
    e_f32[i] = v;
    e_bf[i]  = (__bf16)v;
  }
  __syncthreads();

  // ---- GEMM1: h1 = relu(e @ w1 + b1)   [16 x 512], 32 N-tiles / 8 waves ----
  for (int tt = 0; tt < 4; ++tt) {
    const int n0 = (wv + tt * 8) * 16;
    const float bias = b1[n0 + (lane & 15)];
    v8f acc;
#pragma unroll
    for (int v = 0; v < 8; ++v) acc[v] = bias;
#pragma unroll
    for (int k0 = 0; k0 < HD; k0 += 32) {
      v16bf a = wmma_load_a(e_bf, HD, k0, lane);
      v16bf b = wmma_load_b(w1t, HD, n0, k0, lane);
      acc = wmma_bf16(a, b, acc);
    }
    const int col  = n0 + (lane & 15);
    const int rofs = (lane < 16) ? 0 : 8;
#pragma unroll
    for (int v = 0; v < 8; ++v) {
      float f = acc[v];
      f = f > 0.f ? f : 0.f;
      h1_bf[(v + rofs) * FFD + col] = (__bf16)f;
    }
  }
  __syncthreads();

  // ---- GEMM2: x = h1 @ w2 + b2 + e   [16 x 256], 16 N-tiles / 8 waves ----
  for (int tt = 0; tt < 2; ++tt) {
    const int n0 = (wv + tt * 8) * 16;
    const float bias = b2[n0 + (lane & 15)];
    v8f acc;
#pragma unroll
    for (int v = 0; v < 8; ++v) acc[v] = bias;
#pragma unroll
    for (int k0 = 0; k0 < FFD; k0 += 32) {
      v16bf a = wmma_load_a(h1_bf, FFD, k0, lane);
      v16bf b = wmma_load_b(w2t, FFD, n0, k0, lane);
      acc = wmma_bf16(a, b, acc);
    }
    const int col  = n0 + (lane & 15);
    const int rofs = (lane < 16) ? 0 : 8;
#pragma unroll
    for (int v = 0; v < 8; ++v) {
      const int m = v + rofs;
      x_f32[m * HD + col] = acc[v] + e_f32[m * HD + col];
    }
  }
  __syncthreads();

  // ---- LayerNorm: wave wv handles rows 2wv, 2wv+1 ----
  for (int rr = 0; rr < 2; ++rr) {
    const int m = wv * 2 + rr;
    const float* xr = x_f32 + m * HD;
    float s = 0.f, sq = 0.f;
    for (int j = lane; j < HD; j += 32) { float v = xr[j]; s += v; sq += v * v; }
#pragma unroll
    for (int off = 16; off > 0; off >>= 1) {
      s  += __shfl_xor(s,  off, 32);
      sq += __shfl_xor(sq, off, 32);
    }
    const float mu   = s * (1.f / HD);
    const float var  = sq * (1.f / HD) - mu * mu;
    const float rstd = rsqrtf(var + 1e-5f);
    for (int j = lane; j < HD; j += 32) {
      float hh = (xr[j] - mu) * rstd * ln_g[j] + ln_b[j];
      h_bf[m * HD + j] = (__bf16)hh;
    }
  }
  __syncthreads();

  // ---- GEMM3: k = h @ kp_w   [16 x 256] -> global k_all (fp32) ----
  for (int tt = 0; tt < 2; ++tt) {
    const int n0 = (wv + tt * 8) * 16;
    v8f acc;
#pragma unroll
    for (int v = 0; v < 8; ++v) acc[v] = 0.f;
#pragma unroll
    for (int k0 = 0; k0 < HD; k0 += 32) {
      v16bf a = wmma_load_a(h_bf, HD, k0, lane);
      v16bf b = wmma_load_b(kpt, HD, n0, k0, lane);
      acc = wmma_bf16(a, b, acc);
    }
    const int col  = n0 + (lane & 15);
    const int rofs = (lane < 16) ? 0 : 8;
#pragma unroll
    for (int v = 0; v < 8; ++v)
      k_all[(size_t)(rowbase + v + rofs) * HD + col] = acc[v];
  }
}

// ---------------------------------------------------------------------------
// Delta-rule fast-weight scan, WY/chunked form (exact rewrite of the
// sequential rank-1 recurrence). One block per batch; M (256x256 f32) is
// LDS-resident for all 64 chunks. Per chunk (C=32):
//   P  = M0 @ KN^T           (WMMA, A = fp32 M converted on the fly)
//   S  = KN @ KN^T           (WMMA)
//   dv_t = k_t - P[:,t] - sum_{s<t} S[t,s] dv_s   (unrolled tri-solve, VALU)
//   M += DV^T @ KN           (WMMA, f32 accumulate into LDS)
// LDS: 256KB (M) + 16KB (kn t-major) + 16KB (kn^T j-major)
//      + 16KB (P-then-DV, aliased in place) + 4KB (S) = 315.4 KB  (< 320 KB)
// ---------------------------------------------------------------------------
__global__ void __launch_bounds__(256)
fastweight_scan_wy(const float* __restrict__ k_all, float* __restrict__ r_out) {
  extern __shared__ char smem[];
  float*  M     = (float*)smem;                       // 256 x 256 f32
  __bf16* kn_bf = (__bf16*)(M + 256 * HD);            // CCH x 256  (t-major)
  __bf16* knT   = kn_bf + CCH * HD;                   // 256 x CCH  (j-major)
  __bf16* ptdv  = knT + HD * CCH;                     // 256 x CCH  (P, then DV)
  float*  Sm    = (float*)(ptdv + HD * CCH);          // CCH x CCH f32

  const int tid  = threadIdx.x;
  const int lane = tid & 31;
  const int wv   = tid >> 5;
  const int b    = blockIdx.x;
  const float* kb = k_all + (size_t)b * SL * HD;

  for (int i = tid; i < HD * HD; i += 256) M[i] = 0.f;
  __syncthreads();

  for (int c0 = 0; c0 < SL - 1; c0 += CCH) {
    const int Tc = (SL - 1 - c0 < CCH) ? (SL - 1 - c0) : CCH;

    // ---- normalize k_t; wave wv handles t = wv, wv+8, wv+16, wv+24 ----
#pragma unroll
    for (int tt = 0; tt < CCH / 8; ++tt) {
      const int t = wv + tt * 8;
      if (t < Tc) {
        float kv[8];
        float sq = 0.f;
        const float* kr = kb + (size_t)(c0 + t) * HD + lane * 8;
#pragma unroll
        for (int u = 0; u < 8; ++u) { kv[u] = kr[u]; sq += kv[u] * kv[u]; }
#pragma unroll
        for (int off = 16; off > 0; off >>= 1) sq += __shfl_xor(sq, off, 32);
        const float inv = 1.f / fmaxf(sqrtf(sq), 1e-12f);
#pragma unroll
        for (int u = 0; u < 8; ++u) {
          const int j = lane * 8 + u;
          const __bf16 v = (__bf16)(kv[u] * inv);
          kn_bf[t * HD + j] = v;
          knT[j * CCH + t]  = v;
        }
      } else {
#pragma unroll
        for (int u = 0; u < 8; ++u) {
          const int j = lane * 8 + u;
          kn_bf[t * HD + j] = (__bf16)0.f;
          knT[j * CCH + t]  = (__bf16)0.f;
        }
      }
    }
    __syncthreads();

    // ---- P^T = M0 @ KN^T (256 x 32): 32 tiles -> 4 per wave ----
    for (int tt = 0; tt < 4; ++tt) {
      const int mi = wv * 2 + (tt >> 1);
      const int ni = tt & 1;
      v8f acc;
#pragma unroll
      for (int v = 0; v < 8; ++v) acc[v] = 0.f;
#pragma unroll
      for (int k0 = 0; k0 < HD; k0 += 32) {
        v16bf a = wmma_load_a_f32(M + mi * 16 * HD, HD, k0, lane);
        v16bf bm = wmma_load_b(kn_bf, HD, ni * 16, k0, lane);
        acc = wmma_bf16(a, bm, acc);
      }
      const int n = ni * 16 + (lane & 15);
      const int rofs = (lane < 16) ? 0 : 8;
#pragma unroll
      for (int v = 0; v < 8; ++v)
        ptdv[(mi * 16 + v + rofs) * CCH + n] = (__bf16)acc[v];
    }

    // ---- S = KN @ KN^T (32 x 32): 4 tiles on waves 0..3 ----
    if (wv < 4) {
      const int mi = wv >> 1;
      const int ni = wv & 1;
      v8f acc;
#pragma unroll
      for (int v = 0; v < 8; ++v) acc[v] = 0.f;
#pragma unroll
      for (int k0 = 0; k0 < HD; k0 += 32) {
        v16bf a  = wmma_load_a(kn_bf + mi * 16 * HD, HD, k0, lane);
        v16bf bm = wmma_load_b(kn_bf, HD, ni * 16, k0, lane);
        acc = wmma_bf16(a, bm, acc);
      }
      const int n = ni * 16 + (lane & 15);
      const int rofs = (lane < 16) ? 0 : 8;
#pragma unroll
      for (int v = 0; v < 8; ++v)
        Sm[(mi * 16 + v + rofs) * CCH + n] = acc[v];
    }
    __syncthreads();

    // ---- forward substitution: thread j owns feature column j ----
    {
      float dvreg[CCH];
      const int j = tid;
#pragma unroll
      for (int t = 0; t < CCH; ++t) {
        float acc = 0.f;
        if (t < Tc) {
          acc = kb[(size_t)(c0 + t) * HD + j] - (float)ptdv[j * CCH + t];
#pragma unroll
          for (int s = 0; s < t; ++s) acc -= Sm[t * CCH + s] * dvreg[s];
        }
        dvreg[t] = acc;
        ptdv[j * CCH + t] = (__bf16)acc;   // overwrite P slot with DV, in place
      }
    }
    __syncthreads();

    // ---- M += DV^T @ KN : 256 tiles, 32 per wave, K=32 (one WMMA each) ----
    for (int tt = 0; tt < 32; ++tt) {
      const int mi = wv * 2 + (tt >> 4);
      const int ni = tt & 15;
      v16bf a  = wmma_load_a(ptdv + mi * 16 * CCH, CCH, 0, lane);
      v16bf bm = wmma_load_b(knT, CCH, ni * 16, 0, lane);
      v8f acc;
#pragma unroll
      for (int v = 0; v < 8; ++v) acc[v] = 0.f;
      acc = wmma_bf16(a, bm, acc);
      const int n = ni * 16 + (lane & 15);
      const int rofs = (lane < 16) ? 0 : 8;
#pragma unroll
      for (int v = 0; v < 8; ++v)
        M[(mi * 16 + v + rofs) * HD + n] += acc[v];
    }
    __syncthreads();
  }

  // ---- r = M @ q, q = k_all[:, L-1, :] (unnormalized) ----
  float* qv = Sm;  // reuse 4KB scratch
  qv[tid] = kb[(size_t)(SL - 1) * HD + tid];
  __syncthreads();
  float r = 0.f;
  const float* Mr = M + tid * HD;
#pragma unroll 8
  for (int j = 0; j < HD; ++j) r += Mr[j] * qv[j];
  r_out[b * HD + tid] = r;
}

// ---------------------------------------------------------------------------
// t = r @ rp_w + rp_b (16x256, tiny) -> bf16 for the vocab GEMM
// ---------------------------------------------------------------------------
__global__ void __launch_bounds__(256)
small_proj(const float* __restrict__ r, const float* __restrict__ rp_w,
           const float* __restrict__ rp_b, __bf16* __restrict__ t_bf) {
  const int tid = threadIdx.x;
  for (int idx = tid; idx < NB * HD; idx += 256) {
    const int bb = idx >> 8, n = idx & 255;
    float acc = rp_b[n];
    const float* rr = r + bb * HD;
#pragma unroll 8
    for (int j = 0; j < HD; ++j) acc += rr[j] * rp_w[j * HD + n];
    t_bf[idx] = (__bf16)acc;
  }
}

// ---------------------------------------------------------------------------
// out = t @ out_w + out_b   [16 x 32000], K=256. 8 waves/block, 1 N-tile/wave.
// ---------------------------------------------------------------------------
__global__ void __launch_bounds__(256)
out_gemm(const __bf16* __restrict__ t_bf, const __bf16* __restrict__ owt,
         const float* __restrict__ out_b, float* __restrict__ out) {
  const int lane = threadIdx.x & 31;
  const int wv   = threadIdx.x >> 5;
  const int n0   = (blockIdx.x * 8 + wv) * 16;

  const float bias = out_b[n0 + (lane & 15)];
  v8f acc;
#pragma unroll
  for (int v = 0; v < 8; ++v) acc[v] = bias;
#pragma unroll
  for (int k0 = 0; k0 < HD; k0 += 32) {
    v16bf a = wmma_load_a(t_bf, HD, k0, lane);
    v16bf b = wmma_load_b(owt, HD, n0, k0, lane);
    acc = wmma_bf16(a, b, acc);
  }
  const int col  = n0 + (lane & 15);
  const int rofs = (lane < 16) ? 0 : 8;
#pragma unroll
  for (int v = 0; v < 8; ++v)
    out[(size_t)(v + rofs) * VOC + col] = acc[v];
}

// ---------------------------------------------------------------------------
extern "C" void kernel_launch(void* const* d_in, const int* in_sizes, int n_in,
                              void* d_out, int out_size, void* d_ws, size_t ws_size,
                              hipStream_t stream) {
  const int*   seq   = (const int*)  d_in[0];
  const float* embed = (const float*)d_in[1];
  const float* w1    = (const float*)d_in[2];
  const float* b1    = (const float*)d_in[3];
  const float* w2    = (const float*)d_in[4];
  const float* b2    = (const float*)d_in[5];
  const float* ln_g  = (const float*)d_in[6];
  const float* ln_b  = (const float*)d_in[7];
  const float* kp_w  = (const float*)d_in[8];
  const float* rp_w  = (const float*)d_in[9];
  const float* rp_b  = (const float*)d_in[10];
  const float* out_w = (const float*)d_in[11];
  const float* out_b = (const float*)d_in[12];

  char* ws = (char*)d_ws;
  size_t off = 0;
  __bf16* w1t  = (__bf16*)(ws + off); off += (size_t)FFD * HD * 2;        // (512,256)
  __bf16* w2t  = (__bf16*)(ws + off); off += (size_t)HD * FFD * 2;        // (256,512)
  __bf16* kpt  = (__bf16*)(ws + off); off += (size_t)HD * HD * 2;         // (256,256)
  __bf16* owt  = (__bf16*)(ws + off); off += (size_t)VOC * HD * 2;        // (32000,256)
  float*  kall = (float*)(ws + off);  off += (size_t)NB * SL * HD * 4;    // 33.5 MB
  float*  rbuf = (float*)(ws + off);  off += (size_t)NB * HD * 4;
  __bf16* t_bf = (__bf16*)(ws + off); off += (size_t)NB * HD * 2;

  // 1) weight convert + transpose to bf16 (N,K)
  conv_transpose_bf16<<<512, 256, 0, stream>>>(w1, w1t, HD, FFD);
  conv_transpose_bf16<<<512, 256, 0, stream>>>(w2, w2t, FFD, HD);
  conv_transpose_bf16<<<256, 256, 0, stream>>>(kp_w, kpt, HD, HD);
  conv_transpose_bf16<<<8192, 256, 0, stream>>>(out_w, owt, HD, VOC);

  // 2) fused embed/FFN/LN/k-proj (WMMA), 2048 blocks x 16 rows
  ffn_ln_kproj<<<(NB * SL) / 16, 256, 64 * 1024, stream>>>(
      seq, embed, w1t, b1, w2t, b2, ln_g, ln_b, kpt, kall);

  // 3) WY-chunked delta-rule scan (WMMA), one WGP per batch, M LDS-resident
  size_t scan_lds = (size_t)HD * HD * 4      // M fp32
                  + (size_t)CCH * HD * 2     // kn (t-major)
                  + (size_t)HD * CCH * 2     // kn^T (j-major)
                  + (size_t)HD * CCH * 2     // P / DV (aliased)
                  + (size_t)CCH * CCH * 4;   // S
  fastweight_scan_wy<<<NB, 256, scan_lds, stream>>>(kall, rbuf);

  // 4) tiny r@rp_w projection -> bf16
  small_proj<<<1, 256, 0, stream>>>(rbuf, rp_w, rp_b, t_bf);

  // 5) vocab GEMM (WMMA), 2000 N-tiles
  out_gemm<<<VOC / (16 * 8), 256, 0, stream>>>(t_bf, owt, out_b, (float*)d_out);
}
